// LSTMusic_9938554323552
// MI455X (gfx1250) — matrix operand order repaired and checked by reading, not compiled, not run on hardware
//
#include <hip/hip_runtime.h>

// ---------------- problem constants (from reference) ----------------
#define B_TOT   2048
#define N_TOT   255
#define HIST    64
#define ND      32
#define SD      3
#define H       64
#define INDIM   36          // SD + 1 + ND
#define KPAD    128         // [x(36) | h(64) | bias-one(1) | zero-pad] -> 4 K=32 chunks
#define NGATES  256         // 4*H
#define NGEN    191         // N_TOT - HIST
#define GAPT    192         // N_TOT - HIST + 1
#define TOUT    256         // HIST + NGEN + 1
#define KBIAS   100         // K index whose A value is constant 1.0 (bias row)

// ---------------- launch geometry ----------------
#define WAVES           4
#define TPB             (WAVES * 32)
#define ROWS_PER_WAVE   16
#define ROWS_PER_BLOCK  (WAVES * ROWS_PER_WAVE)
#define NBLK            (B_TOT / ROWS_PER_BLOCK)

// ---------------- dynamic LDS layout (bytes) ----------------
#define WBUF_BYTES  (64 * 32 * 16 * 2)               // 65536
#define OFF_WDF     WBUF_BYTES                        // Wd B-operands, kc=1..3
#define WDF_BYTES   (3 * 32 * 16 * 2)                 // 3072
#define OFF_DP      (OFF_WDF + WDF_BYTES)             // per-wave dp bounce
#define DP_BYTES    (WAVES * 16 * 4)                  // 256
#define OFF_XH      (OFF_DP + DP_BYTES)               // 16B aligned
#define XH_BYTES    (WAVES * ROWS_PER_WAVE * KPAD * 2)
#define SMEM_BYTES  (OFF_XH + XH_BYTES)               // ~83 KB < 320 KB/WGP

typedef __attribute__((ext_vector_type(16))) __bf16        v16bf;
typedef __attribute__((ext_vector_type(8)))  float         v8f;
typedef __attribute__((ext_vector_type(4)))  unsigned int  v4u;

union BFrag { v16bf v; v4u q[2]; };

__device__ __forceinline__ unsigned short f2bf(float f) {
  union { float f; unsigned int u; } v; v.f = f;
  return (unsigned short)((v.u + 0x8000u) >> 16);
}
__device__ __forceinline__ unsigned int pack2bf(float a, float b) {
  return (unsigned int)f2bf(a) | ((unsigned int)f2bf(b) << 16);
}

#if __has_builtin(__builtin_amdgcn_rcpf)
__device__ __forceinline__ float frcp(float x) { return __builtin_amdgcn_rcpf(x); }
#else
__device__ __forceinline__ float frcp(float x) { return 1.0f / x; }
#endif
__device__ __forceinline__ float fsig(float x) { return frcp(1.0f + __expf(-x)); }
#if __has_builtin(__builtin_amdgcn_tanhf)
__device__ __forceinline__ float ftanh(float x) { return __builtin_amdgcn_tanhf(x); }
#else
__device__ __forceinline__ float ftanh(float x) { return 2.0f * fsig(2.0f * x) - 1.0f; }
#endif

// Load a 16x32 bf16 A fragment (frag fi covers K in [32*fi, 32*fi+32)) from xh.
__device__ __forceinline__ v16bf loadA(const unsigned short* xh, int rlane, int half, int fi) {
  const unsigned short* p = xh + rlane * KPAD + (fi << 5) + (half << 4);
  BFrag u;
  u.q[0] = *(const v4u*)(p);        // K = 32*fi + 16*half + 0..7
  u.q[1] = *(const v4u*)(p + 8);    // K = 32*fi + 16*half + 8..15
  return u.v;
}
// Load one pre-swizzled B fragment (32 lanes * 16 bf16, contiguous per lane).
__device__ __forceinline__ v16bf loadB(const unsigned short* base, int slot, int lane) {
  const unsigned short* p = base + (slot * 32 + lane) * 16;
  BFrag u;
  u.q[0] = *(const v4u*)p;
  u.q[1] = *(const v4u*)(p + 8);
  return u.v;
}

__global__ __launch_bounds__(TPB, 1)
void lstmusic_kernel(const float* __restrict__ noise,     // (B, 255, 32)
                     const float* __restrict__ hist_x,    // (B, 64, 3)
                     const float* __restrict__ gap,       // (B, 192, 2)
                     const float* __restrict__ W,         // (36, 256)
                     const float* __restrict__ U,         // (64, 256)
                     const float* __restrict__ bvec,      // (256,)
                     const float* __restrict__ Wd,        // (64,)
                     const float* __restrict__ bd,        // (1,)
                     float* __restrict__ out)             // (B, 256, 3)
{
  extern __shared__ char smem[];
  unsigned short* wbuf   = (unsigned short*)smem;
  unsigned short* wdf    = (unsigned short*)(smem + OFF_WDF);
  float*          dpbuf  = (float*)(smem + OFF_DP);
  unsigned short* xh_all = (unsigned short*)(smem + OFF_XH);

  const int tid  = threadIdx.x;
  const int wave = tid >> 5;
  const int lane = tid & 31;

  // ---------------- one-time weight staging into LDS ----------------
  // Combined weights [W;U;bias;0] (KPAD x 256), per-lane WMMA B-operand layout:
  // fragment (nt,kc): lane l holds K = 32*kc + 16*(l/16) + j, N = 16*nt + l%16.
  for (int s = tid; s < 64 * 32; s += TPB) {
    const int fragIdx = s >> 5;
    const int fl      = s & 31;
    const int nt      = fragIdx >> 2;
    const int kc      = fragIdx & 3;
    const int col     = (nt << 4) + (fl & 15);
    const int khalf   = (fl >> 4) << 4;
    unsigned short* dst = wbuf + s * 16;
#pragma unroll
    for (int j = 0; j < 16; ++j) {
      const int K = (kc << 5) + khalf + j;
      float v = 0.0f;
      if (K < INDIM)            v = W[(size_t)K * NGATES + col];
      else if (K < INDIM + H)   v = U[(size_t)(K - INDIM) * NGATES + col];
      else if (K == KBIAS)      v = bvec[col];        // bias row (A supplies 1.0)
      dst[j] = f2bf(v);
    }
  }
  // Wd as a B operand in column 0, K chunks 1..3 (rows 36..99); everything else 0.
  for (int s = tid; s < 3 * 32; s += TPB) {
    const int kci   = (s >> 5) + 1;
    const int fl    = s & 31;
    const int col   = fl & 15;
    const int khalf = (fl >> 4) << 4;
    unsigned short* dst = wdf + s * 16;
#pragma unroll
    for (int j = 0; j < 16; ++j) {
      const int K = (kci << 5) + khalf + j;
      float v = (col == 0 && K >= INDIM && K < INDIM + H) ? Wd[K - INDIM] : 0.0f;
      dst[j] = f2bf(v);
    }
  }
  // xh: zeros, except constant 1.0 at K=KBIAS of every row (bias input).
  for (int i = tid; i < WAVES * ROWS_PER_WAVE * KPAD; i += TPB)
    xh_all[i] = ((i & (KPAD - 1)) == KBIAS) ? (unsigned short)0x3F80 : (unsigned short)0;
  __syncthreads();

  // ---------------- per-wave recurrent state ----------------
  unsigned short* xh  = xh_all + wave * (ROWS_PER_WAVE * KPAD);
  float*          dpw = dpbuf + wave * 16;
  const int rowBase = blockIdx.x * ROWS_PER_BLOCK + wave * ROWS_PER_WAVE;
  const int rlane   = lane & 15;
  const int half    = lane >> 4;
  const int row     = rowBase + rlane;
  const float bdreg = bd[0];                      // uniform scalar load

  float dist = 0.0f, p0 = 0.0f, p1 = 0.0f, p2 = 0.0f;  // valid in lanes 0..15
  v8f cst[4];                                          // cell state, C/D layout
#pragma unroll
  for (int j = 0; j < 4; ++j)
#pragma unroll
    for (int e = 0; e < 8; ++e) cst[j][e] = 0.0f;

  // Noise staging split across lane halves; loads pipelined one step ahead.
  const int noff = half << 4;                      // 0 or 16
  const float* nbase = noise + (size_t)row * N_TOT * ND + noff;
  float4 nzv[4];
  {
    const float4* np = (const float4*)nbase;       // t = 0
#pragma unroll
    for (int q = 0; q < 4; ++q) nzv[q] = np[q];
  }

#pragma unroll 1
  for (int t = 0; t < N_TOT; ++t) {
    v16bf A2, A3;                                  // h-side frags reused dp -> gates
    const bool gen = (t >= HIST);
    if (gen) {
      // ---- dp pre-activation = h_prev . Wd via 3 WMMAs on column 0 ----
      v16bf Ah1 = loadA(xh, rlane, half, 1);       // K 32..63 (stale x -> B rows 0)
      A2        = loadA(xh, rlane, half, 2);       // K 64..95
      A3        = loadA(xh, rlane, half, 3);       // K 96..127 (bias row -> B row 0)
      v16bf wd0 = loadB(wdf, 0, lane);
      v16bf wd1 = loadB(wdf, 1, lane);
      v16bf wd2 = loadB(wdf, 2, lane);
      v8f zd = {};
      zd = __builtin_amdgcn_wmma_f32_16x16x32_bf16(false, Ah1, false, wd0, (short)0, zd, false, false);
      zd = __builtin_amdgcn_wmma_f32_16x16x32_bf16(false, A2,  false, wd1, (short)0, zd, false, false);
      zd = __builtin_amdgcn_wmma_f32_16x16x32_bf16(false, A3,  false, wd2, (short)0, zd, false, false);
      if (rlane == 0) {                            // lanes 0 & 16 hold column 0
#pragma unroll
        for (int e = 0; e < 8; ++e)
          dpw[e + (half << 3)] = zd[e];            // raw pre-activation; tanh done
      }                                            // by the row-owner lane below
    }

    // ---- per-row features f[0..3] + dist/cumsum/output (row-owner lanes) ----
    if (lane < 16) {
      float f0, f1, f2;
      if (!gen) {
        const float* hx = hist_x + ((size_t)row * HIST + t) * SD;
        f0 = hx[0]; f1 = hx[1]; f2 = hx[2];
        dist += f2;                                // inclusive cumsum of dpitch
        float* op = out + ((size_t)row * TOUT + t) * SD;   // out[:, :64] = hist_x
        op[0] = f0; op[1] = f1; op[2] = f2;
      } else {
        const float dp = ftanh(dpw[rlane] + bdreg) * 24.0f;  // 1 tanh per lane
        const int tg = t - HIST;
        const float* gp = gap + ((size_t)row * GAPT + tg) * 2;
        f0 = gp[0]; f1 = gp[1]; f2 = dp;
        dist += dp;
        p0 += f0; p1 += f1; p2 += f2;              // cumsum of gen_seq
        float* op = out + ((size_t)row * TOUT + HIST + tg) * SD;
        op[0] = p0; op[1] = p1; op[2] = p2;
      }
      *(unsigned int*)(xh + rlane * KPAD + 0) = pack2bf(f0, f1);
      *(unsigned int*)(xh + rlane * KPAD + 2) = pack2bf(f2, dist);
    }
    // ---- noise features K=4..35, both halves in parallel ----
    {
      const float* nf = (const float*)nzv;
#pragma unroll
      for (int m = 0; m < 8; ++m)
        *(unsigned int*)(xh + rlane * KPAD + 4 + noff + 2 * m) = pack2bf(nf[2 * m], nf[2 * m + 1]);
    }
    // prefetch next step's noise (off the critical path)
    if (t + 1 < N_TOT) {
      const float4* np = (const float4*)(nbase + (size_t)(t + 1) * ND);
#pragma unroll
      for (int q = 0; q < 4; ++q) nzv[q] = np[q];
    }

    // ---- A fragments for the gate GEMM ----
    v16bf A[4];
    A[0] = loadA(xh, rlane, half, 0);
    A[1] = loadA(xh, rlane, half, 1);              // reload: fresh x K32..35
    if (gen) { A[2] = A2; A[3] = A3; }
    else     { A[2] = loadA(xh, rlane, half, 2); A[3] = loadA(xh, rlane, half, 3); }

    // ---- gates: z = [x|h|1] @ [W;U;b] (bias folded in), 64 WMMAs ----
    // Explicit per-kc fragment preloads: give the scheduler 4 distinct B
    // registers so ds_load latency overlaps the WMMA pipe (peeled dscnt waits).
#pragma unroll
    for (int j = 0; j < 4; ++j) {                  // hidden slice (16 units)
      v8f zi = {}, zf = {}, zg = {}, zo = {};
#pragma unroll
      for (int kc = 0; kc < 4; ++kc) {
        const v16bf wi = loadB(wbuf, (0  + j) * 4 + kc, lane);
        const v16bf wf = loadB(wbuf, (4  + j) * 4 + kc, lane);
        const v16bf wg = loadB(wbuf, (8  + j) * 4 + kc, lane);
        const v16bf wo = loadB(wbuf, (12 + j) * 4 + kc, lane);
        zi = __builtin_amdgcn_wmma_f32_16x16x32_bf16(false, A[kc], false, wi, (short)0, zi, false, false);
        zf = __builtin_amdgcn_wmma_f32_16x16x32_bf16(false, A[kc], false, wf, (short)0, zf, false, false);
        zg = __builtin_amdgcn_wmma_f32_16x16x32_bf16(false, A[kc], false, wg, (short)0, zg, false, false);
        zo = __builtin_amdgcn_wmma_f32_16x16x32_bf16(false, A[kc], false, wo, (short)0, zo, false, false);
      }
      const int n = (j << 4) + rlane;              // hidden index, per lane
#pragma unroll
      for (int e = 0; e < 8; ++e) {
        const float iv = fsig (zi[e]);
        const float fv = fsig (zf[e]);
        const float gv = ftanh(zg[e]);
        const float ov = fsig (zo[e]);
        const float cn = fv * cst[j][e] + iv * gv;
        cst[j][e] = cn;
        const float hv = ov * ftanh(cn);
        const int Mr = e + (half << 3);            // C/D layout row
        xh[Mr * KPAD + INDIM + n] = f2bf(hv);      // h back to LDS for next step
      }
    }
  }

  // ---- final x_last (out index 255): dp from final h via the same 3 WMMAs ----
  {
    v16bf Ah1 = loadA(xh, rlane, half, 1);
    v16bf Ah2 = loadA(xh, rlane, half, 2);
    v16bf Ah3 = loadA(xh, rlane, half, 3);
    v16bf wd0 = loadB(wdf, 0, lane);
    v16bf wd1 = loadB(wdf, 1, lane);
    v16bf wd2 = loadB(wdf, 2, lane);
    v8f zd = {};
    zd = __builtin_amdgcn_wmma_f32_16x16x32_bf16(false, Ah1, false, wd0, (short)0, zd, false, false);
    zd = __builtin_amdgcn_wmma_f32_16x16x32_bf16(false, Ah2, false, wd1, (short)0, zd, false, false);
    zd = __builtin_amdgcn_wmma_f32_16x16x32_bf16(false, Ah3, false, wd2, (short)0, zd, false, false);
    if (rlane == 0) {
#pragma unroll
      for (int e = 0; e < 8; ++e)
        dpw[e + (half << 3)] = zd[e];
    }
    if (lane < 16) {
      const float dp = ftanh(dpw[rlane] + bdreg) * 24.0f;
      const float* gp = gap + ((size_t)row * GAPT + NGEN) * 2;
      p0 += gp[0]; p1 += gp[1]; p2 += dp;
      float* op = out + ((size_t)row * TOUT + (TOUT - 1)) * SD;
      op[0] = p0; op[1] = p1; op[2] = p2;
    }
  }
}

extern "C" void kernel_launch(void* const* d_in, const int* in_sizes, int n_in,
                              void* d_out, int out_size, void* d_ws, size_t ws_size,
                              hipStream_t stream) {
  const float* noise  = (const float*)d_in[0];
  const float* hist_x = (const float*)d_in[1];
  const float* gap    = (const float*)d_in[2];
  const float* W      = (const float*)d_in[3];
  const float* U      = (const float*)d_in[4];
  const float* b      = (const float*)d_in[5];
  const float* Wd     = (const float*)d_in[6];
  const float* bd     = (const float*)d_in[7];
  float* out = (float*)d_out;
  (void)in_sizes; (void)n_in; (void)out_size; (void)d_ws; (void)ws_size;

  lstmusic_kernel<<<dim3(NBLK), dim3(TPB), SMEM_BYTES, stream>>>(
      noise, hist_x, gap, W, U, b, Wd, bd, out);
}
